// multi_chan_linop_22170621182453
// MI455X (gfx1250) — matrix operand order repaired and checked by reading, not compiled, not run on hardware
//
#include <hip/hip_runtime.h>

// ---------------------------------------------------------------------------
// Type-2 NUFFT (exact nonuniform DFT) for NC=8, NX=NY=128, K=16384.
//   ksp[c,k] = sum_x sum_y exp(-i*2pi/128 * trj[k,0]*(x-64)) * img[c,x,y]
//                         * exp(-i*2pi/128 * trj[k,1]*(y-64))
// Heavy contraction (over x): bf16 WMMA with hi/lo split precision (3
// products per real GEMM -> fp32-class accuracy at ~2.7x the f32-WMMA MAC
// rate).
//  * Each block: one channel x TWO 16-row k-tiles. The img B-operand is
//    k-independent, so both k-tiles share every B fragment -> L2 B-traffic
//    halves (1 GB -> 512 MB) and per-chunk the mix is 8 global b128 +
//    16 ds b128 + 24 WMMA.
//  * phase_x tiles: built once per block cooperatively into LDS (8x dedup
//    across the 8 y-tile waves), fed to the matrix pipe via ds_load_b128.
//  * img B-tiles: fp32->bf16 hi/lo split precomputed ONCE by a pre-kernel
//    into d_ws, transposed [c][y][x] so each lane's fragment is a 32B
//    contiguous read (2x global_load_b128, zero conversion VALU in-loop).
// Light contraction (over y): VALU + lane shuffles + LDS combine.
// ---------------------------------------------------------------------------

constexpr int kNC = 8;
constexpr int kNX = 128;
constexpr int kNY = 128;
constexpr int kK  = 16384;
constexpr int kPitch = 136;   // bf16 elements per phase row: 128 + 8 pad
                              // row stride = 272B -> LDS bank stride 4
constexpr float kAlpha = -6.283185307179586f / 128.0f; // -2*pi/N (N==128)

typedef __attribute__((ext_vector_type(16))) __bf16 v16bf;
typedef __attribute__((ext_vector_type(8)))  __bf16 v8bf;
typedef __attribute__((ext_vector_type(2)))  __bf16 v2bf;
typedef __attribute__((ext_vector_type(8)))  float  v8f;

__device__ __forceinline__ void split2(float v, __bf16 &hi, __bf16 &lo) {
    float h = (float)(__bf16)v;      // round-to-nearest bf16, back to f32
    hi = (__bf16)h;
    lo = (__bf16)(v - h);            // residual
}

// Assemble a 16-element bf16 fragment from two 16B runs at p and p+stride.
__device__ __forceinline__ v16bf frag2x8(const __bf16* p, int stride) {
    v8bf a = *(const v8bf*)p;
    v8bf b = *(const v8bf*)(p + stride);
    return __builtin_shufflevector(a, b, 0, 1, 2, 3, 4, 5, 6, 7,
                                         8, 9, 10, 11, 12, 13, 14, 15);
}

#define WMMA_BF16(A, B, C) \
    __builtin_amdgcn_wmma_f32_16x16x32_bf16(false, (A), false, (B), (short)0, (C), false, false)

// ---------------------------------------------------------------------------
// Pre-kernel: split fp32 image into bf16 hi/lo, transposed to [c][y][x].
// ---------------------------------------------------------------------------
__global__ __launch_bounds__(256)
void presplit_kernel(const float* __restrict__ re, const float* __restrict__ im,
                     __bf16* __restrict__ hR, __bf16* __restrict__ lR,
                     __bf16* __restrict__ hI, __bf16* __restrict__ lI) {
    int t = blockIdx.x * 256 + threadIdx.x;   // over NC*NX*NY, y fastest
    int y = t & (kNY - 1);
    int x = (t >> 7) & (kNX - 1);
    int c = t >> 14;
    float vr = re[t];                         // coalesced along y
    float vi = im[t];
    int o = (c * kNY + y) * kNX + x;          // transposed [c][y][x]
    __bf16 h, l;
    split2(vr, h, l); hR[o] = h; lR[o] = l;
    split2(vi, h, l); hI[o] = h; lI[o] = l;
}

// ---------------------------------------------------------------------------
// Main kernel. USE_WS: B operands from pre-split bf16 (fast path) or from
// fp32 global with in-loop split (fallback if d_ws is too small).
// Block = (channel, 32 k-rows); wave = one 16-column y-tile.
// ---------------------------------------------------------------------------
template <bool USE_WS>
__global__ __launch_bounds__(256)
void nudft_wmma_kernel(const float* __restrict__ img_real,
                       const float* __restrict__ img_imag,
                       const float* __restrict__ trj,
                       const __bf16* __restrict__ bHR,
                       const __bf16* __restrict__ bLR,
                       const __bf16* __restrict__ bHI,
                       const __bf16* __restrict__ bLI,
                       float2* __restrict__ out) {
    const int lane  = threadIdx.x & 31;
    const int wv    = threadIdx.x >> 5;       // y-tile (0..7), 16 columns each
    const int ybase = wv * 16;
    const int ch    = blockIdx.x >> 9;        // channel (K/32 = 512 blocks)
    const int kbase = (blockIdx.x & 511) * 32;

    __shared__ __attribute__((aligned(16))) __bf16 sCosH[32 * kPitch];
    __shared__ __attribute__((aligned(16))) __bf16 sCosL[32 * kPitch];
    __shared__ __attribute__((aligned(16))) __bf16 sSinH[32 * kPitch];
    __shared__ __attribute__((aligned(16))) __bf16 sSinL[32 * kPitch];
    __shared__ float2 partial[8][32];

    // ---- Cooperative phase_x tile build: 32 k-rows x 128 x, split bf16 ----
    // 2048 element-pairs over 256 threads: 8 iters, 2 sincos each.
    #pragma unroll
    for (int i = 0; i < 8; ++i) {
        int idx = threadIdx.x + i * 256;      // pair index 0..2047
        int kk  = idx >> 6;                   // 0..31
        int x0  = (idx & 63) * 2;             // even x
        float t0 = trj[2 * (kbase + kk)];
        float s0, c0, s1, c1;
        __sincosf(kAlpha * t0 * (float)(x0 - kNX / 2), &s0, &c0);
        __sincosf(kAlpha * t0 * (float)(x0 + 1 - kNX / 2), &s1, &c1);
        __bf16 h0, l0, h1, l1;
        int off = kk * kPitch + x0;
        split2(c0, h0, l0); split2(c1, h1, l1);
        *(v2bf*)(sCosH + off) = v2bf{h0, h1};
        *(v2bf*)(sCosL + off) = v2bf{l0, l1};
        split2(s0, h0, l0); split2(s1, h1, l1);
        *(v2bf*)(sSinH + off) = v2bf{h0, h1};
        *(v2bf*)(sSinL + off) = v2bf{l0, l1};
    }
    __syncthreads();

    const int halfsel = (lane & 16) >> 1;     // 0 for lanes 0-15, 8 for 16-31
    const int ycol    = ybase + (lane & 15);  // B-matrix column for this lane

    v8f accRR[2] = {v8f{}, v8f{}};
    v8f accII[2] = {v8f{}, v8f{}};
    v8f accRI[2] = {v8f{}, v8f{}};
    v8f accIR[2] = {v8f{}, v8f{}};

    #pragma unroll
    for (int chunk = 0; chunk < 4; ++chunk) {        // x in chunks of 32
        // --- B operand (shared by both k-tiles): contiguous k order (same as A),
        //     element e -> K = e + (lane&16) -> 16 consecutive x values.
        v16bf Irh, Irl, Iih, Iil;
        if constexpr (USE_WS) {
            int boff = (ch * kNY + ycol) * kNX + chunk * 32 + (lane & 16);
            Irh = frag2x8(bHR + boff, 8);     // 2x global_load_b128 each
            Irl = frag2x8(bLR + boff, 8);
            Iih = frag2x8(bHI + boff, 8);
            Iil = frag2x8(bLI + boff, 8);
        } else {
            #pragma unroll
            for (int e = 0; e < 16; ++e) {
                int xb = chunk * 32 + e + (lane & 16);
                int idx = (ch * kNX + xb) * kNY + ycol;
                __bf16 h, l;
                split2(img_real[idx], h, l); Irh[e] = h; Irl[e] = l;
                split2(img_imag[idx], h, l); Iih[e] = h; Iil[e] = l;
            }
        }

        // --- Two k-tiles, each: A fragments from LDS + 12 WMMAs ---
        int run0 = chunk * 32 + (lane & 16);  // A: 16 consecutive x, K = e + (lane&16)
        #pragma unroll
        for (int t = 0; t < 2; ++t) {
            int rowOff = ((lane & 15) + 16 * t) * kPitch;
            v16bf Prh = frag2x8(sCosH + rowOff + run0, 8);
            v16bf Prl = frag2x8(sCosL + rowOff + run0, 8);
            v16bf Pih = frag2x8(sSinH + rowOff + run0, 8);
            v16bf Pil = frag2x8(sSinL + rowOff + run0, 8);

            // Complex GEMM (4 real accumulators), 3-term hi/lo split each.
            accRR[t] = WMMA_BF16(Prh, Irh, accRR[t]);
            accRR[t] = WMMA_BF16(Prh, Irl, accRR[t]);
            accRR[t] = WMMA_BF16(Prl, Irh, accRR[t]);

            accII[t] = WMMA_BF16(Pih, Iih, accII[t]);
            accII[t] = WMMA_BF16(Pih, Iil, accII[t]);
            accII[t] = WMMA_BF16(Pil, Iih, accII[t]);

            accRI[t] = WMMA_BF16(Prh, Iih, accRI[t]);
            accRI[t] = WMMA_BF16(Prh, Iil, accRI[t]);
            accRI[t] = WMMA_BF16(Prl, Iih, accRI[t]);

            accIR[t] = WMMA_BF16(Pih, Irh, accIR[t]);
            accIR[t] = WMMA_BF16(Pih, Irl, accIR[t]);
            accIR[t] = WMMA_BF16(Pil, Irh, accIR[t]);
        }
    }

    // Stage 2: multiply by phase_y and reduce over y, per k-tile.
    // C layout: VGPR r, lanes 0-15 -> (M=r, N=lane); lanes 16-31 -> (M=r+8).
    const float yv = (float)(ycol - kNY / 2);
    #pragma unroll
    for (int t = 0; t < 2; ++t) {
        v8f Ar, Ai;
        #pragma unroll
        for (int r = 0; r < 8; ++r) {
            Ar[r] = accRR[t][r] - accII[t][r];
            Ai[r] = accRI[t][r] + accIR[t][r];
        }
        const float t1 = trj[2 * (kbase + 16 * t + (lane & 15)) + 1];
        #pragma unroll
        for (int r = 0; r < 8; ++r) {
            float t1k = __shfl(t1, r + halfsel, 32); // trj row for M = r(+8)
            float s2, c2;
            __sincosf(kAlpha * t1k * yv, &s2, &c2);
            float pr = Ar[r] * c2 - Ai[r] * s2;
            float pi = Ar[r] * s2 + Ai[r] * c2;
            // reduce across the 16 lanes of each half (same M, y tile cols)
            #pragma unroll
            for (int m = 1; m <= 8; m <<= 1) {
                pr += __shfl_xor(pr, m, 32);
                pi += __shfl_xor(pi, m, 32);
            }
            if ((lane & 15) == 0) {
                partial[wv][16 * t + r + halfsel] = make_float2(pr, pi);
            }
        }
    }
    __syncthreads();

    // Combine the 8 y-tiles (one per wave) and emit complex64 output.
    if (threadIdx.x < 32) {
        float sr = 0.0f, si = 0.0f;
        #pragma unroll
        for (int w2 = 0; w2 < 8; ++w2) {
            sr += partial[w2][threadIdx.x].x;
            si += partial[w2][threadIdx.x].y;
        }
        out[ch * kK + kbase + threadIdx.x] = make_float2(sr, si);
    }
}

extern "C" void kernel_launch(void* const* d_in, const int* in_sizes, int n_in,
                              void* d_out, int out_size, void* d_ws, size_t ws_size,
                              hipStream_t stream) {
    const float* img_real = (const float*)d_in[0];  // (NC, NX, NY) f32
    const float* img_imag = (const float*)d_in[1];  // (NC, NX, NY) f32
    const float* trj      = (const float*)d_in[2];  // (K, 2) f32
    float2* out = (float2*)d_out;                   // (NC, K) complex64

    const size_t n    = (size_t)kNC * kNX * kNY;    // 131072 elements
    const size_t need = 4 * n * sizeof(__bf16);     // 1 MB of scratch

    dim3 grid(kNC * (kK / 32));   // 4096 blocks: one (channel, 2 k-tiles) each
    dim3 block(256);              // 8 wave32s: one 16-column y-tile each

    if (ws_size >= need) {
        __bf16* hR = (__bf16*)d_ws;
        __bf16* lR = hR + n;
        __bf16* hI = lR + n;
        __bf16* lI = hI + n;
        presplit_kernel<<<dim3((unsigned)(n / 256)), block, 0, stream>>>(
            img_real, img_imag, hR, lR, hI, lI);
        nudft_wmma_kernel<true><<<grid, block, 0, stream>>>(
            img_real, img_imag, trj, hR, lR, hI, lI, out);
    } else {
        nudft_wmma_kernel<false><<<grid, block, 0, stream>>>(
            img_real, img_imag, trj, nullptr, nullptr, nullptr, nullptr, out);
    }

    (void)in_sizes; (void)n_in; (void)out_size;
}
